// GNN_BM_37907381354586
// MI455X (gfx1250) — matrix operand order repaired and checked by reading, compiled, not run on hardware
//
#include <hip/hip_runtime.h>

// ---------------------------------------------------------------------------
// 6-layer GAT on a 48^3 grid (MI455X / gfx1250, wave32).
//  - GEMMs via v_wmma_f32_16x16x32_bf16 (fp32 accum), W staged in LDS as bf16,
//    K/OUT compile-time so WMMA tile loops fully unroll (no branching).
//  - inter-layer activations stored as bf16 (converted once at the attention
//    writeback) so GEMM A-fragments load with zero conversion VALU and half
//    the bytes; everything is L2-resident on the 192MB L2.
//  - attention done gather-style per destination node (grid structure known),
//    deterministic, no atomics
//  - replication-pad BC fused into the next layer's GEMM row indexing
// ---------------------------------------------------------------------------

#define GRD 48
#define NN  (GRD * GRD * GRD)      // 110592 nodes (divisible by 128 and 8)

typedef __bf16 v16bf __attribute__((ext_vector_type(16)));
typedef __bf16 v8bf  __attribute__((ext_vector_type(8)));
typedef float  v8f   __attribute__((ext_vector_type(8)));

static __device__ __forceinline__ __bf16 f2bf(float f) {
    unsigned u = __builtin_bit_cast(unsigned, f);
    u += 0x7fffu + ((u >> 16) & 1u);            // round-to-nearest-even
    unsigned short s = (unsigned short)(u >> 16);
    return __builtin_bit_cast(__bf16, s);
}

static __device__ __forceinline__ void pack4(v16bf& f, int o, float4 a) {
    f[o + 0] = f2bf(a.x); f[o + 1] = f2bf(a.y);
    f[o + 2] = f2bf(a.z); f[o + 3] = f2bf(a.w);
}

// ---------------------------------------------------------------------------
// GEMM: Hout[N x OUT] = Xin[N x K] @ W[K x OUT], bf16 WMMA, fp32 accumulate.
// BF16IN: Xin rows already bf16 (direct fragment loads). Otherwise fp32 + cvt.
// applyBC!=0: row r of Xin read from the coord-clamped node (replication pad).
// Block: 256 threads (8 waves); block covers 128 rows x OUT cols.
// ---------------------------------------------------------------------------
template<int K, int OUT, bool BF16IN>
__global__ __launch_bounds__(256) void gemm_bf16_wmma(
    const void* __restrict__ Xin, const float* __restrict__ W,
    float* __restrict__ Hout, int applyBC)
{
    constexpr int Kpad   = (K + 31) & ~31;
    constexpr int NT     = (OUT + 15) >> 4;
    constexpr int OUTpad = NT << 4;
    constexpr int NKC    = Kpad >> 5;

    __shared__ __align__(64) __bf16 sW[OUTpad * Kpad];   // W^T bf16: sW[n*Kpad+k]

    const int tid = threadIdx.x;
    #pragma unroll 4
    for (int idx = tid; idx < OUTpad * Kpad; idx += 256) {
        int n = idx / Kpad, k = idx - n * Kpad;
        float v = (n < OUT && k < K) ? W[(size_t)k * OUT + n] : 0.0f;
        sW[idx] = f2bf(v);
    }
    __syncthreads();

    const int wave  = tid >> 5;
    const int lane  = tid & 31;
    const int l16   = lane & 15;     // M row (A frag) / N col (B frag)
    const int khalf = lane >> 4;     // K-half selector

    // A-fragment source row (boundary-condition clamp fused in)
    int row  = blockIdx.x * 128 + wave * 16 + l16;
    int srow = row;
    if (applyBC) {
        int zz = row % GRD, yy = (row / GRD) % GRD, xx = row / (GRD * GRD);
        xx = min(max(xx, 1), GRD - 2);
        yy = min(max(yy, 1), GRD - 2);
        zz = min(max(zz, 1), GRD - 2);
        srow = (xx * GRD + yy) * GRD + zz;
    }

    v8f acc[NT] = {};

    #pragma unroll
    for (int kc = 0; kc < NKC; ++kc) {
        const int kb = kc * 32 + khalf * 16;  // A layout: slots0-15 = kb..kb+15 (contiguous)
        v16bf afrag;
        if constexpr (BF16IN) {
            const __bf16* xrow = (const __bf16*)Xin + (size_t)srow * K;
            v8bf lo = *(const v8bf*)(xrow + kb);        // 16B aligned
            v8bf hi = *(const v8bf*)(xrow + kb + 8);
            afrag = __builtin_shufflevector(lo, hi, 0, 1, 2, 3, 4, 5, 6, 7,
                                                    8, 9, 10, 11, 12, 13, 14, 15);
        } else {
            const float* xrow = (const float*)Xin + (size_t)srow * K;
            afrag = v16bf{};
            if ((kc + 1) * 32 <= K) {         // compile-time after unroll
                pack4(afrag, 0,  *(const float4*)(xrow + kb));
                pack4(afrag, 4,  *(const float4*)(xrow + kb + 4));
                pack4(afrag, 8,  *(const float4*)(xrow + kb + 8));
                pack4(afrag, 12, *(const float4*)(xrow + kb + 12));
            } else {                          // tail (layer 0, K=10): guarded scalar
                #pragma unroll
                for (int i = 0; i < 8; ++i) {
                    int k0 = kb + i, k1 = kb + 8 + i;
                    afrag[i]     = f2bf(k0 < K ? xrow[k0] : 0.0f);
                    afrag[8 + i] = f2bf(k1 < K ? xrow[k1] : 0.0f);
                }
            }
        }
        #pragma unroll
        for (int t = 0; t < NT; ++t) {
            // B layout: lane=N col, 16 contiguous K at kc*32 + khalf*16 (32B aligned)
            const v16bf bfrag =
                *(const v16bf*)(&sW[(size_t)(t * 16 + l16) * Kpad + kc * 32 + khalf * 16]);
            acc[t] = __builtin_amdgcn_wmma_f32_16x16x32_bf16(
                false, afrag, false, bfrag, (short)0, acc[t], false, false);
        }
    }

    // C/D layout: lane half selects M base (0 or 8), VGPR v = M offset, lane&15 = N
    const int rbase = blockIdx.x * 128 + wave * 16 + khalf * 8;
    #pragma unroll
    for (int t = 0; t < NT; ++t) {
        int col = t * 16 + l16;
        if (col < OUT) {
            #pragma unroll
            for (int v = 0; v < 8; ++v)
                Hout[(size_t)(rbase + v) * OUT + col] = acc[t][v];
        }
    }
}

// ---------------------------------------------------------------------------
// es/ed: per (node, head) attention logit halves.  es[n,h] = <h[n,h,:], a_s[h,:]>
// ---------------------------------------------------------------------------
__global__ __launch_bounds__(256) void esed_kernel(
    const float* __restrict__ h, const float* __restrict__ a_s,
    const float* __restrict__ a_d, float* __restrict__ es, float* __restrict__ ed,
    int H, int C, int total)
{
    int t = blockIdx.x * 256 + threadIdx.x;
    if (t >= total) return;
    int n = t / H, hd = t - n * H;
    const float* row = h + (size_t)n * H * C + hd * C;
    float s = 0.0f, d = 0.0f;
    for (int c = 0; c < C; ++c) {
        float v = row[c];
        s += v * a_s[hd * C + c];
        d += v * a_d[hd * C + c];
    }
    es[t] = s;
    ed[t] = d;
}

// ---------------------------------------------------------------------------
// Attention aggregation (gather form): per dst node, <=7 incoming sources
// (6 grid neighbors that exist + self loop). Softmax over incoming edges,
// weighted sum of h[src], +bias, optional ReLU.  One wave per node.
// BF16OUT: store activations as bf16 for the next layer's GEMM A-fragments.
// ---------------------------------------------------------------------------
template<bool BF16OUT>
__global__ __launch_bounds__(256) void gat_aggregate(
    const float* __restrict__ h, const float* __restrict__ es,
    const float* __restrict__ ed, const float* __restrict__ bias,
    void* __restrict__ out, int H, int C, int doRelu)
{
    __shared__ int   s_src[8][8];
    __shared__ int   s_cnt[8];
    __shared__ float s_wt[8][4][8];

    const int tid  = threadIdx.x;
    const int w    = tid >> 5;
    const int lane = tid & 31;
    const int node = blockIdx.x * 8 + w;

    if (lane == 0) {
        int zz = node % GRD, yy = (node / GRD) % GRD, xx = node / (GRD * GRD);
        int c = 0;
        if (xx > 0)       s_src[w][c++] = node - GRD * GRD;
        if (xx < GRD - 1) s_src[w][c++] = node + GRD * GRD;
        if (yy > 0)       s_src[w][c++] = node - GRD;
        if (yy < GRD - 1) s_src[w][c++] = node + GRD;
        if (zz > 0)       s_src[w][c++] = node - 1;
        if (zz < GRD - 1) s_src[w][c++] = node + 1;
        s_src[w][c++] = node;                 // self loop
        s_cnt[w] = c;
    }
    __syncthreads();

    const int cnt = s_cnt[w];
    if (lane < H) {
        float edv = ed[(size_t)node * H + lane];
        float e[8];
        float m = -1e30f;
        for (int k = 0; k < cnt; ++k) {
            float v = es[(size_t)s_src[w][k] * H + lane] + edv;
            v = v > 0.0f ? v : 0.2f * v;      // leaky_relu(0.2)
            e[k] = v;
            m = fmaxf(m, v);
        }
        float ssum = 0.0f;
        for (int k = 0; k < cnt; ++k) { e[k] = __expf(e[k] - m); ssum += e[k]; }
        float inv = 1.0f / (ssum + 1e-16f);
        for (int k = 0; k < cnt; ++k) s_wt[w][lane][k] = e[k] * inv;
    }
    __syncthreads();

    const int F = H * C;
    for (int f = lane; f < F; f += 32) {
        int head = f / C;
        float acc = 0.0f;
        for (int k = 0; k < cnt; ++k)
            acc += s_wt[w][head][k] * h[(size_t)s_src[w][k] * F + f];
        float v = acc + bias[f];
        if (doRelu) v = fmaxf(v, 0.0f);
        if constexpr (BF16OUT)
            ((__bf16*)out)[(size_t)node * F + f] = f2bf(v);
        else
            ((float*)out)[(size_t)node * F + f] = v;
    }
}

// ---------------------------------------------------------------------------
// Final: crop interior (46^3), transpose to (T=2, 5, 46,46,46), abs on c in {0,4}
// ---------------------------------------------------------------------------
__global__ __launch_bounds__(256) void finalize_kernel(
    const float* __restrict__ h, float* __restrict__ y, int total)
{
    int idx = blockIdx.x * 256 + threadIdx.x;
    if (idx >= total) return;
    const int V = GRD - 2, V3 = V * V * V;          // 46, 97336
    int t   = idx / (5 * V3);
    int rem = idx - t * 5 * V3;
    int c   = rem / V3;
    int r2  = rem - c * V3;
    int X = r2 / (V * V);
    int Y = (r2 / V) % V;
    int Z = r2 % V;
    int n = ((X + 1) * GRD + (Y + 1)) * GRD + (Z + 1);
    float v = h[(size_t)n * 10 + t * 5 + c];
    if (c == 0 || c == 4) v = fabsf(v);
    y[idx] = v;
}

// ---------------------------------------------------------------------------
extern "C" void kernel_launch(void* const* d_in, const int* in_sizes, int n_in,
                              void* d_out, int out_size, void* d_ws, size_t ws_size,
                              hipStream_t stream)
{
    (void)in_sizes; (void)n_in; (void)out_size; (void)ws_size;

    const float* x = (const float*)d_in[0];
    // d_in[1] = edge_index: structured grid, regenerated implicitly in-kernel.
    const float *Wl[6], *asl[6], *adl[6], *bl[6];
    for (int i = 0; i < 6; ++i) {
        Wl[i]  = (const float*)d_in[2 + 4 * i];
        asl[i] = (const float*)d_in[3 + 4 * i];
        adl[i] = (const float*)d_in[4 + 4 * i];
        bl[i]  = (const float*)d_in[5 + 4 * i];
    }

    // Workspace layout (~90 MB; L2-resident on MI455X's 192 MB L2)
    float*  Hbuf = (float*)d_ws;             // [NN * 128] f32   GEMM output h
    __bf16* Xbf  = (__bf16*)(Hbuf + (size_t)NN * 128); // [NN * 128] bf16 activations
    float*  Xf32 = (float*)Xbf;              // layer-5 output view (NN*10 f32)
    float*  es   = (float*)(Xbf + (size_t)NN * 128);   // [NN * 4]
    float*  ed   = es + (size_t)NN * 4;                // [NN * 4]

    for (int l = 0; l < 6; ++l) {
        const int H = (l == 5) ? 1  : 4;
        const int C = (l == 5) ? 10 : 32;

        if (l == 0)
            gemm_bf16_wmma<10, 128, false><<<NN / 128, 256, 0, stream>>>(x, Wl[0], Hbuf, 0);
        else if (l < 5)
            gemm_bf16_wmma<128, 128, true><<<NN / 128, 256, 0, stream>>>(Xbf, Wl[l], Hbuf, 1);
        else
            gemm_bf16_wmma<128, 10, true><<<NN / 128, 256, 0, stream>>>(Xbf, Wl[5], Hbuf, 1);

        const int tot = NN * H;
        esed_kernel<<<(tot + 255) / 256, 256, 0, stream>>>(Hbuf, asl[l], adl[l], es, ed, H, C, tot);

        if (l < 5)
            gat_aggregate<true><<<NN / 8, 256, 0, stream>>>(Hbuf, es, ed, bl[l], Xbf, H, C, 1);
        else
            gat_aggregate<false><<<NN / 8, 256, 0, stream>>>(Hbuf, es, ed, bl[l], Xf32, H, C, 0);
    }

    const int total = 2 * 5 * 46 * 46 * 46;   // 973360
    finalize_kernel<<<(total + 255) / 256, 256, 0, stream>>>(Xf32, (float*)d_out, total);
}
